// MultiScaleFeatureExtractor_49435073577253
// MI455X (gfx1250) — compile-verified, matrix operand order checked by reading, not run on hardware
//
#include <hip/hip_runtime.h>
#include <hip/hip_bf16.h>

// ---------------------------------------------------------------------------
// MultiScaleFeatureExtractor fused kernel for MI455X (gfx1250, wave32, WMMA)
//
// out[b,h,s,d] = (sum_n softmax_s(px@Wslice/T)[n,s] * fx[n,d])
//                / (sum_n softmax_s(...)[n,s] + 0.01)
// Single pass over tokens; numerator kept in WMMA accumulator registers.
// ---------------------------------------------------------------------------

#define BATCH   2
#define NTOK    65536
#define CDIM    256
#define HEADS   8
#define DIMH    64
#define SLICES  64
#define INNER   512

#define TILE    64                         // tokens per inner iteration
#define CHUNKS  32                         // token chunks per (b,h)
#define TOKS_PER_BLOCK (NTOK / CHUNKS)     // 2048

typedef __attribute__((ext_vector_type(16))) __bf16        v16bf;
typedef __attribute__((ext_vector_type(8)))  __bf16        v8bf;
typedef __attribute__((ext_vector_type(8)))  float         v8f;
typedef __attribute__((ext_vector_type(8)))  unsigned short v8us;

__device__ __forceinline__ unsigned short f2bf(float f) {
    union { float f; unsigned int u; } v; v.f = f;
    unsigned int u = v.u;
    unsigned int r = u + 0x7FFFu + ((u >> 16) & 1u);   // round-to-nearest-even
    return (unsigned short)(r >> 16);
}
__device__ __forceinline__ float bf2f(unsigned short s) {
    union { unsigned int u; float f; } v; v.u = ((unsigned int)s) << 16;
    return v.f;
}

// Load a 16x32 bf16 fragment (A or transposed-B) from LDS, row-major, ld elems.
// ISA 16-bit A layout: lanes 0-15 hold row M=lane, K = {0..7, 16..23};
// lanes 16-31 hold row M=lane-16, K = {8..15, 24..31}. Two b128 LDS loads.
__device__ __forceinline__ v16bf load_frag(const unsigned short* base, int ld, int lane) {
    int hl = lane >> 4;
    int r  = lane & 15;
    const unsigned short* p = base + r * ld + hl * 8;
    v8bf lo = __builtin_bit_cast(v8bf, *(const v8us*)(p));
    v8bf hi = __builtin_bit_cast(v8bf, *(const v8us*)(p + 16));
    v16bf f;
#pragma unroll
    for (int e = 0; e < 8; ++e) { f[e] = lo[e]; f[e + 8] = hi[e]; }
    return f;
}

__global__ void __launch_bounds__(256)
fused_slice_kernel(const float* __restrict__ x,
                   const float* __restrict__ Wx,  const float* __restrict__ bx,
                   const float* __restrict__ Wfx, const float* __restrict__ bfx,
                   const float* __restrict__ Wslice, const float* __restrict__ bslice,
                   const float* __restrict__ temperature,
                   float* __restrict__ numer_g, float* __restrict__ denom_g)
{
    extern __shared__ char smem[];
    unsigned short* xs   = (unsigned short*)smem;          // [64][256] x tile (bf16)
    unsigned short* WxT  = xs   + TILE  * CDIM;            // [64][256] Wx head slice, transposed
    unsigned short* WfxT = WxT  + DIMH  * CDIM;            // [64][256]
    unsigned short* WsT  = WfxT + DIMH  * CDIM;            // [64][64]  Wslice transposed
    unsigned short* px   = WsT  + SLICES * DIMH;           // [64][64]  px tile (n-major)
    unsigned short* fxT  = px   + TILE  * DIMH;            // [64][64]  fx tile (d-major)
    unsigned short* wT   = fxT  + DIMH  * TILE;            // [64][64]  softmax weights (s-major)
    float*          sp   = (float*)(wT + SLICES * TILE);   // [64][64]  slice logits (f32)
    float*          denom = sp + TILE * SLICES;            // [64]

    const int tid  = threadIdx.x;
    const int wave = tid >> 5;
    const int lane = tid & 31;
    const int hl   = lane >> 4;
    const int lcol = lane & 15;

    // h fastest so the 8 head-blocks sharing an x chunk are co-resident (L2 reuse)
    const int bid   = blockIdx.x;
    const int h     = bid % HEADS;
    const int chunk = (bid / HEADS) % CHUNKS;
    const int b     = bid / (HEADS * CHUNKS);
    const int tok0  = chunk * TOKS_PER_BLOCK;

    // ---- stage weights (transposed: WT[j][k] = W[k][h*64+j]) ----
    for (int i = tid; i < DIMH * CDIM; i += 256) {
        int j = i >> 8, k = i & 255;
        WxT[i]  = f2bf(Wx [k * INNER + h * DIMH + j]);
        WfxT[i] = f2bf(Wfx[k * INNER + h * DIMH + j]);
    }
    for (int i = tid; i < SLICES * DIMH; i += 256) {
        int j = i >> 6, k = i & 63;
        WsT[i] = f2bf(Wslice[k * SLICES + j]);
    }
    if (tid < SLICES) denom[tid] = 0.0f;

    float tv = temperature[h];
    tv = fminf(fmaxf(tv, 0.5f), 5.0f);
    const float inv_temp = 1.0f / tv;

    // 4x4 grid of 16x16 output tiles; wave owns tiles t0, t0+1
    const int t0  = wave * 2;
    const int ti0 = t0 >> 2,        tj0 = t0 & 3;
    const int ti1 = (t0 + 1) >> 2,  tj1 = (t0 + 1) & 3;

    v8f acc0 = {}; v8f acc1 = {};   // numerator accumulators, live across the loop

    __syncthreads();

    for (int it = 0; it < TOKS_PER_BLOCK / TILE; ++it) {
        const int base_tok = tok0 + it * TILE;

        // ---- 1) load x tile -> bf16 LDS ----
        {
            const float* xg = x + ((size_t)b * NTOK + base_tok) * CDIM;
            for (int i = tid * 4; i < TILE * CDIM; i += 256 * 4) {
                float4 v = *(const float4*)(xg + i);
                xs[i + 0] = f2bf(v.x); xs[i + 1] = f2bf(v.y);
                xs[i + 2] = f2bf(v.z); xs[i + 3] = f2bf(v.w);
            }
        }
        __syncthreads();

        // ---- 2) GEMM1: px = xs@Wxh + bx ; fxT = (xs@Wfxh + bfx)^T ----
#pragma unroll
        for (int q = 0; q < 2; ++q) {
            const int ti = q ? ti1 : ti0, tj = q ? tj1 : tj0;
            const int rb = ti * 16, cb = tj * 16;
            v8f c = {}; v8f cf = {};
#pragma unroll
            for (int kk = 0; kk < 8; ++kk) {
                v16bf a  = load_frag(xs   + rb * CDIM + kk * 32, CDIM, lane);
                v16bf bw = load_frag(WxT  + cb * CDIM + kk * 32, CDIM, lane);
                v16bf bf = load_frag(WfxT + cb * CDIM + kk * 32, CDIM, lane);
                c  = __builtin_amdgcn_wmma_f32_16x16x32_bf16(false, a, false, bw, (short)0, c,  false, false);
                cf = __builtin_amdgcn_wmma_f32_16x16x32_bf16(false, a, false, bf, (short)0, cf, false, false);
            }
            const float biasx  = bx [h * DIMH + cb + lcol];
            const float biasfx = bfx[h * DIMH + cb + lcol];
#pragma unroll
            for (int i = 0; i < 8; ++i) {
                int row = rb + i + hl * 8;
                int col = cb + lcol;
                px [row * DIMH + col] = f2bf(c[i]  + biasx);
                fxT[col * TILE + row] = f2bf(cf[i] + biasfx);
            }
        }
        __syncthreads();

        // ---- 3) GEMM2: sp = (px@Wslice + bslice) / temp ----
#pragma unroll
        for (int q = 0; q < 2; ++q) {
            const int ti = q ? ti1 : ti0, tj = q ? tj1 : tj0;
            const int rb = ti * 16, cb = tj * 16;
            v8f c = {};
#pragma unroll
            for (int kk = 0; kk < 2; ++kk) {
                v16bf a  = load_frag(px  + rb * DIMH + kk * 32, DIMH, lane);
                v16bf bw = load_frag(WsT + cb * DIMH + kk * 32, DIMH, lane);
                c = __builtin_amdgcn_wmma_f32_16x16x32_bf16(false, a, false, bw, (short)0, c, false, false);
            }
            const float biass = bslice[cb + lcol];
#pragma unroll
            for (int i = 0; i < 8; ++i)
                sp[(rb + i + hl * 8) * SLICES + cb + lcol] = (c[i] + biass) * inv_temp;
        }
        __syncthreads();

        // ---- 4) softmax over slices per token; write wT[s][n] (bf16) ----
        if (tid < TILE) {
            const int n = tid;
            float mx = -3.4e38f;
            for (int s = 0; s < SLICES; ++s) mx = fmaxf(mx, sp[n * SLICES + s]);
            float sum = 0.0f;
            for (int s = 0; s < SLICES; ++s) sum += __expf(sp[n * SLICES + s] - mx);
            const float inv = 1.0f / sum;
            for (int s = 0; s < SLICES; ++s)
                wT[s * TILE + n] = f2bf(__expf(sp[n * SLICES + s] - mx) * inv);
        }
        __syncthreads();

        // ---- 5) GEMM3: numer += wT @ fx ; denom[s] += sum_n w[n,s] ----
#pragma unroll
        for (int kk = 0; kk < 2; ++kk) {
            v16bf a0 = load_frag(wT  + ti0 * 16 * TILE + kk * 32, TILE, lane);
            v16bf b0 = load_frag(fxT + tj0 * 16 * TILE + kk * 32, TILE, lane);
            acc0 = __builtin_amdgcn_wmma_f32_16x16x32_bf16(false, a0, false, b0, (short)0, acc0, false, false);
            v16bf a1 = load_frag(wT  + ti1 * 16 * TILE + kk * 32, TILE, lane);
            v16bf b1 = load_frag(fxT + tj1 * 16 * TILE + kk * 32, TILE, lane);
            acc1 = __builtin_amdgcn_wmma_f32_16x16x32_bf16(false, a1, false, b1, (short)0, acc1, false, false);
        }
        if (tid < SLICES) {
            const int s = tid; float sum = 0.0f;
#pragma unroll 8
            for (int n = 0; n < TILE; ++n) sum += bf2f(wT[s * TILE + n]);
            denom[s] += sum;    // single writer per s
        }
        __syncthreads();
    }

    // ---- epilogue: accumulate partials into global workspace ----
    float* numer_bh = numer_g + ((size_t)(b * HEADS + h)) * SLICES * DIMH;
#pragma unroll
    for (int q = 0; q < 2; ++q) {
        const int ti = q ? ti1 : ti0, tj = q ? tj1 : tj0;
        v8f c = q ? acc1 : acc0;
#pragma unroll
        for (int i = 0; i < 8; ++i) {
            int s = ti * 16 + i + hl * 8;
            int d = tj * 16 + lcol;
            atomicAdd(&numer_bh[s * DIMH + d], c[i]);
        }
    }
    if (tid < SLICES)
        atomicAdd(&denom_g[(b * HEADS + h) * SLICES + tid], denom[tid]);
}

__global__ void zero_ws_kernel(float* numer, float* denom, int n_numer, int n_denom) {
    int i = blockIdx.x * blockDim.x + threadIdx.x;
    if (i < n_numer) numer[i] = 0.0f;
    if (i < n_denom) denom[i] = 0.0f;
}

__global__ void finalize_kernel(const float* __restrict__ numer,
                                const float* __restrict__ denom,
                                float* __restrict__ out, int total) {
    int i = blockIdx.x * blockDim.x + threadIdx.x;
    if (i < total) out[i] = numer[i] / (denom[i / DIMH] + 0.01f);
}

extern "C" void kernel_launch(void* const* d_in, const int* in_sizes, int n_in,
                              void* d_out, int out_size, void* d_ws, size_t ws_size,
                              hipStream_t stream) {
    const float* x      = (const float*)d_in[0];
    const float* Wx     = (const float*)d_in[1];
    const float* bx     = (const float*)d_in[2];
    const float* Wfx    = (const float*)d_in[3];
    const float* bfx    = (const float*)d_in[4];
    const float* Wslice = (const float*)d_in[5];
    const float* bslice = (const float*)d_in[6];
    const float* temp   = (const float*)d_in[7];
    float* out   = (float*)d_out;
    float* numer = (float*)d_ws;                               // 65536 f32
    float* denom = numer + BATCH * HEADS * SLICES * DIMH;      // 1024 f32

    const int n_numer = BATCH * HEADS * SLICES * DIMH;
    const int n_denom = BATCH * HEADS * SLICES;

    zero_ws_kernel<<<(n_numer + 255) / 256, 256, 0, stream>>>(numer, denom, n_numer, n_denom);

    // LDS: 3x(64x256) + 4x(64x64) bf16 + 64x64 f32 + 64 f32 = 147,712 bytes
    const size_t shmem =
        (size_t)(TILE * CDIM + 2 * DIMH * CDIM + SLICES * DIMH +
                 TILE * DIMH + DIMH * TILE + SLICES * TILE) * 2 +
        (size_t)(TILE * SLICES) * 4 + SLICES * 4;

    fused_slice_kernel<<<BATCH * HEADS * CHUNKS, 256, shmem, stream>>>(
        x, Wx, bx, Wfx, bfx, Wslice, bslice, temp, numer, denom);

    finalize_kernel<<<(n_numer + 255) / 256, 256, 0, stream>>>(numer, denom, out, n_numer);
}
